// Attention_41042707480856
// MI455X (gfx1250) — matrix operand order repaired and checked by reading, not compiled, not run on hardware
//
#include <hip/hip_runtime.h>
#include <stdint.h>

// L=8, N=32, V=512, C=256, MV=512
#define LL 8
#define NN 32
#define VV 512
#define CC 256
#define MV 512
#define ROWS (LL*NN*VV)   // 131072, same for P and M since V==MV

typedef __attribute__((ext_vector_type(16))) __bf16 v16bf;
typedef __attribute__((ext_vector_type(8)))  float  v8f;

static __device__ __forceinline__ __bf16 f2bf(float f) {
  return (__bf16)f;                       // native cvt (RNE) on gfx1250
}
static __device__ __forceinline__ uint32_t pack2(float a, float b) {
  union { __bf16 b[2]; uint32_t u; } r;
  r.b[0] = (__bf16)a; r.b[1] = (__bf16)b;
  return r.u;
}
// 16 consecutive bf16 (unused; B fragments use frag_split): one 32B chunk
static __device__ __forceinline__ v16bf frag_c32(const __bf16* p) {
  union { v16bf v; uint4 u[2]; } r;
  r.u[0] = ((const uint4*)p)[0];
  r.u[1] = ((const uint4*)p)[1];
  return r.v;
}
// A-fragment pattern: two non-adjacent 8-element (16B) chunks
static __device__ __forceinline__ v16bf frag_split(const __bf16* p0, const __bf16* p1) {
  union { v16bf v; uint4 u[2]; } r;
  r.u[0] = *(const uint4*)p0;
  r.u[1] = *(const uint4*)p1;
  return r.v;
}
static __device__ __forceinline__ v8f wmma_bf16(v16bf a, v16bf b, v8f c) {
  return __builtin_amdgcn_wmma_f32_16x16x32_bf16(false, a, false, b, (short)0, c, false, false);
}

// gfx1250 async global->LDS copy (ASYNCcnt-tracked, no VGPR staging).
// lds_off: wave-relative LDS byte offset (low 32 bits of generic shared ptr,
// valid because the LDS aperture base has zero low bits).
static __device__ __forceinline__ void async_copy_b128(uint32_t lds_off, const void* gaddr) {
  asm volatile("global_load_async_to_lds_b128 %0, %1, off"
               :: "v"(lds_off), "v"(gaddr) : "memory");
}
static __device__ __forceinline__ void wait_async() {
  asm volatile("s_wait_asynccnt 0x0" ::: "memory");
}
static __device__ __forceinline__ uint32_t lds_off32(const void* p) {
  return (uint32_t)(uintptr_t)p;
}

// ---------------------------------------------------------------------------
__global__ void zero_att(float* att) {
  att[blockIdx.x * 256 + threadIdx.x] = 0.0f;
}

// ---------------------------------------------------------------------------
// Projection: Y[row, d] = X[row, :] . W[d, :] + b[d], output bf16.
// grid = (ROWS/128, 2 col-halves, 3 projections), block = 256 (8 waves).
__global__ __launch_bounds__(256) void proj_kernel(
    const float* __restrict__ P,  const float* __restrict__ M,
    const float* __restrict__ Wk, const float* __restrict__ bk,
    const float* __restrict__ Wq, const float* __restrict__ bq,
    const float* __restrict__ Wv, const float* __restrict__ bv,
    __bf16* __restrict__ Qb, __bf16* __restrict__ Kb, __bf16* __restrict__ Vb)
{
  __shared__ __bf16 w_lds[128 * 256];   // 64 KB: this col-half of W, bf16

  const float* X; const float* W; const float* B; __bf16* Y;
  const int z = blockIdx.z;
  if      (z == 0) { X = P; W = Wk; B = bk; Y = Qb; }
  else if (z == 1) { X = M; W = Wq; B = bq; Y = Kb; }
  else             { X = M; W = Wv; B = bv; Y = Vb; }

  const int tid = threadIdx.x;
  const int colBase = blockIdx.y * 128;

  // Convert this col-half of W (rows colBase..+128, k=0..255) fp32 -> bf16 LDS.
  for (int i = 0; i < 32; ++i) {
    int idx4 = tid + i * 256;          // 8192 float4 total
    int row  = idx4 >> 6;              // /64 float4 per row
    int kq   = idx4 & 63;
    float4 w4 = *(const float4*)(W + (size_t)(colBase + row) * CC + kq * 4);
    uint2 pk; pk.x = pack2(w4.x, w4.y); pk.y = pack2(w4.z, w4.w);
    *(uint2*)(&w_lds[row * 256 + kq * 4]) = pk;
  }
  __syncthreads();

  const int lane = tid & 31, wv = tid >> 5;
  const int hf = lane >> 4, lo = lane & 15;
  const size_t rowg = (size_t)blockIdx.x * 128 + wv * 16 + lo;  // this lane's A row
  const float* xrow = X + rowg * CC;

  v8f acc[8];
  for (int nt = 0; nt < 8; ++nt)
    for (int e = 0; e < 8; ++e) acc[nt][e] = 0.0f;

  for (int ks = 0; ks < 8; ++ks) {                // K = 256 in steps of 32
    const int k0 = ks * 32 + hf * 8;
    float4 a0 = *(const float4*)(xrow + k0);
    float4 a1 = *(const float4*)(xrow + k0 + 4);
    float4 a2 = *(const float4*)(xrow + k0 + 16);
    float4 a3 = *(const float4*)(xrow + k0 + 20);
    v16bf afr;
    afr[0]=f2bf(a0.x); afr[1]=f2bf(a0.y); afr[2]=f2bf(a0.z); afr[3]=f2bf(a0.w);
    afr[4]=f2bf(a1.x); afr[5]=f2bf(a1.y); afr[6]=f2bf(a1.z); afr[7]=f2bf(a1.w);
    afr[8]=f2bf(a2.x); afr[9]=f2bf(a2.y); afr[10]=f2bf(a2.z); afr[11]=f2bf(a2.w);
    afr[12]=f2bf(a3.x); afr[13]=f2bf(a3.y); afr[14]=f2bf(a3.z); afr[15]=f2bf(a3.w);
    for (int nt = 0; nt < 8; ++nt) {
      v16bf bfr = frag_split(&w_lds[(nt * 16 + lo) * 256 + ks * 32 + hf * 8], &w_lds[(nt * 16 + lo) * 256 + ks * 32 + hf * 8 + 16]);
      acc[nt] = wmma_bf16(afr, bfr, acc[nt]);
    }
  }
  // bias + store bf16 (D layout: lane = col, element e = row e + 8*hf)
  for (int nt = 0; nt < 8; ++nt) {
    const int col = colBase + nt * 16 + lo;
    const float bias = B[col];
    size_t base = ((size_t)blockIdx.x * 128 + wv * 16 + hf * 8) * CC + col;
    for (int e = 0; e < 8; ++e)
      Y[base + (size_t)e * CC] = f2bf(acc[nt][e] + bias);
  }
}

// ---------------------------------------------------------------------------
// Attention: grid = (V/128, L*N), block = 256 (8 waves x 16 Q rows).
__global__ __launch_bounds__(256) void attn_kernel(
    const __bf16* __restrict__ Qb, const __bf16* __restrict__ Kb,
    const __bf16* __restrict__ Vb, const int* __restrict__ mask,
    float* __restrict__ outp, float* __restrict__ att)
{
  __shared__ __bf16 q_lds[128 * 256];   // 64 KB  Q tile, row-major
  __shared__ __bf16 k_lds[64 * 256];    // 32 KB  K chunk, row-major
  __shared__ __bf16 v_lds[256 * 64];    // 32 KB  V chunk, transposed [c][m]
  __shared__ __bf16 p_lds[8 * 16 * 32]; // 8 KB   per-wave prob tiles (C/D -> A relayout)
  __shared__ float  maskf[512];
  __shared__ float  att_lds[512];

  const int tid = threadIdx.x;
  const int ln  = blockIdx.y;           // l*32 + n
  const int n   = ln & (NN - 1);
  const int vbase = blockIdx.x * 128;
  const size_t rowBase = (size_t)ln * VV;

  for (int m = tid; m < 512; m += 256) {
    maskf[m]   = (mask[n * MV + m] != 0) ? 1.0f : 0.0f;
    att_lds[m] = 0.0f;
  }
  { // Q tile -> LDS via async copy (64 KB, 4096 x b128)
    const char* src = (const char*)(Qb + (rowBase + vbase) * CC);
    const uint32_t dst = lds_off32(q_lds);
    for (int i = tid; i < 4096; i += 256)
      async_copy_b128(dst + i * 16, src + (size_t)i * 16);
  }
  wait_async();
  __syncthreads();

  const int lane = tid & 31, wv = tid >> 5;
  const int hf = lane >> 4, lo = lane & 15;
  const int qrow = (wv * 16 + lo) * 256;          // LDS row offset of this lane's Q row

  float rowm[8], rowl[8];
  for (int e = 0; e < 8; ++e) { rowm[e] = -1e30f; rowl[e] = 0.0f; }

  // ---------------- Pass 1: online row max / sum(exp) ----------------
  for (int mc = 0; mc < 8; ++mc) {
    __syncthreads();
    { const char* src = (const char*)(Kb + (rowBase + mc * 64) * CC);
      const uint32_t dst = lds_off32(k_lds);
      for (int i = tid; i < 2048; i += 256)
        async_copy_b128(dst + i * 16, src + (size_t)i * 16); }
    wait_async();
    __syncthreads();
    if (mc + 1 < 8)   // warm L2/GL1 for the next chunk while we compute
      __builtin_prefetch(Kb + (rowBase + (mc + 1) * 64) * CC + tid * 64, 0, 0);

    v8f acc[4];
    for (int nt = 0; nt < 4; ++nt)
      for (int e = 0; e < 8; ++e) acc[nt][e] = 0.0f;
    for (int ks = 0; ks < 8; ++ks) {               // A-frag loaded once per ks
      v16bf afr = frag_split(&q_lds[qrow + ks * 32 + hf * 8],
                             &q_lds[qrow + ks * 32 + hf * 8 + 16]);
      for (int nt = 0; nt < 4; ++nt) {
        v16bf bfr = frag_split(&k_lds[(nt * 16 + lo) * 256 + ks * 32 + hf * 8], &k_lds[(nt * 16 + lo) * 256 + ks * 32 + hf * 8 + 16]);
        acc[nt] = wmma_bf16(afr, bfr, acc[nt]);
      }
    }
    for (int nt = 0; nt < 4; ++nt) {
      const float flag = maskf[mc * 64 + nt * 16 + lo];
      for (int e = 0; e < 8; ++e) {
        float s = (flag != 0.0f) ? acc[nt][e] : -10.0f;
        float tm = s;                              // row max over 16 cols
        tm = fmaxf(tm, __shfl_xor(tm, 1, 32));
        tm = fmaxf(tm, __shfl_xor(tm, 2, 32));
        tm = fmaxf(tm, __shfl_xor(tm, 4, 32));
        tm = fmaxf(tm, __shfl_xor(tm, 8, 32));
        float nm = fmaxf(rowm[e], tm);
        float pe = __expf(s - nm);
        pe += __shfl_xor(pe, 1, 32);
        pe += __shfl_xor(pe, 2, 32);
        pe += __shfl_xor(pe, 4, 32);
        pe += __shfl_xor(pe, 8, 32);
        rowl[e] = rowl[e] * __expf(rowm[e] - nm) + pe;
        rowm[e] = nm;
      }
    }
  }
  float rinv[8];
  for (int e = 0; e < 8; ++e) rinv[e] = 1.0f / rowl[e];

  // ---------------- Pass 2: p = exp(s-m)/l ; att mean ; O = p @ V ----------------
  v8f o[16];
  for (int ct = 0; ct < 16; ++ct)
    for (int e = 0; e < 8; ++e) o[ct][e] = 0.0f;

  for (int mc = 0; mc < 8; ++mc) {
    __syncthreads();
    { const char* src = (const char*)(Kb + (rowBase + mc * 64) * CC);
      const uint32_t dst = lds_off32(k_lds);
      for (int i = tid; i < 2048; i += 256)
        async_copy_b128(dst + i * 16, src + (size_t)i * 16); }
    { // V chunk: manual copy (transposes during the store)
      const uint4* src = (const uint4*)(Vb + (rowBase + mc * 64) * CC);
      for (int i = tid; i < 2048; i += 256) {
        uint4 d = src[i];
        int ml = i >> 5;                 // m within chunk (32 uint4 per row)
        int c0 = (i & 31) * 8;
        const __bf16* db = (const __bf16*)&d;
        for (int j = 0; j < 8; ++j) v_lds[(c0 + j) * 64 + ml] = db[j];
      } }
    wait_async();
    __syncthreads();

    for (int msub = 0; msub < 2; ++msub) {         // 32 m-columns at a time
      v8f acc[2];
      for (int nt2 = 0; nt2 < 2; ++nt2)
        for (int e = 0; e < 8; ++e) acc[nt2][e] = 0.0f;
      for (int ks = 0; ks < 8; ++ks) {
        v16bf afr = frag_split(&q_lds[qrow + ks * 32 + hf * 8],
                               &q_lds[qrow + ks * 32 + hf * 8 + 16]);
        for (int nt2 = 0; nt2 < 2; ++nt2) {
          const int nt = msub * 2 + nt2;
          v16bf bfr = frag_split(&k_lds[(nt * 16 + lo) * 256 + ks * 32 + hf * 8], &k_lds[(nt * 16 + lo) * 256 + ks * 32 + hf * 8 + 16]);
          acc[nt2] = wmma_bf16(afr, bfr, acc[nt2]);
        }
      }
      for (int nt2 = 0; nt2 < 2; ++nt2) {
        const int nt = msub * 2 + nt2;
        const float flag = maskf[mc * 64 + nt * 16 + lo];
        float asum = 0.0f;
        for (int e = 0; e < 8; ++e) {
          float s = (flag != 0.0f) ? acc[nt2][e] : -10.0f;
          float p = __expf(s - rowm[e]) * rinv[e];
          asum += p;
          p_lds[wv * 512 + (e + 8 * hf) * 32 + nt2 * 16 + lo] = f2bf(p);
        }
        atomicAdd(&att_lds[mc * 64 + nt * 16 + lo], asum);
      }
      // relayout probs C/D -> A fragment (within-wave LDS round trip; ds in-order)
      v16bf pfr = frag_split(&p_lds[wv * 512 + lo * 32 + hf * 8],
                             &p_lds[wv * 512 + lo * 32 + 16 + hf * 8]);
      for (int ct = 0; ct < 16; ++ct) {
        v16bf bfr = frag_split(&v_lds[(ct * 16 + lo) * 64 + msub * 32 + hf * 8], &v_lds[(ct * 16 + lo) * 64 + msub * 32 + hf * 8 + 16]);
        o[ct] = wmma_bf16(pfr, bfr, o[ct]);
      }
    }
  }

  // out[l, n*V + v, c] fp32
  {
    size_t outBase = (rowBase + vbase + wv * 16 + hf * 8) * CC;
    for (int ct = 0; ct < 16; ++ct) {
      const int c = ct * 16 + lo;
      for (int e = 0; e < 8; ++e)
        outp[outBase + (size_t)e * CC + c] = o[ct][e];
    }
  }
  __syncthreads();
  for (int m = tid; m < 512; m += 256)
    atomicAdd(&att[n * MV + m], att_lds[m] * (1.0f / (float)(LL * VV)));
}

// ---------------------------------------------------------------------------
extern "C" void kernel_launch(void* const* d_in, const int* in_sizes, int n_in,
                              void* d_out, int out_size, void* d_ws, size_t ws_size,
                              hipStream_t stream) {
  (void)in_sizes; (void)n_in; (void)out_size; (void)ws_size;
  const float* P    = (const float*)d_in[0];
  const float* M    = (const float*)d_in[1];
  const int*   mask = (const int*)d_in[2];   // jax bool -> integer per harness rules
  const float* Wk   = (const float*)d_in[3];
  const float* bk   = (const float*)d_in[4];
  const float* Wq   = (const float*)d_in[5];
  const float* bq   = (const float*)d_in[6];
  const float* Wv   = (const float*)d_in[7];
  const float* bv   = (const float*)d_in[8];

  float* outp = (float*)d_out;
  float* att  = outp + (size_t)LL * NN * VV * CC;   // tuple: (out, att) concatenated

  __bf16* Qb = (__bf16*)d_ws;                       // 64 MB each, bf16
  __bf16* Kb = Qb + (size_t)ROWS * CC;
  __bf16* Vb = Kb + (size_t)ROWS * CC;

  zero_att<<<dim3(64), dim3(256), 0, stream>>>(att);
  proj_kernel<<<dim3(ROWS / 128, 2, 3), dim3(256), 0, stream>>>(
      P, M, Wk, bk, Wq, bq, Wv, bv, Qb, Kb, Vb);
  attn_kernel<<<dim3(VV / 128, LL * NN), dim3(256), 0, stream>>>(
      Qb, Kb, Vb, mask, outp, att);
}
